// BestModel4_51556787421860
// MI455X (gfx1250) — matrix operand order repaired and checked by reading, not compiled, not run on hardware
//
#include <hip/hip_runtime.h>
#include <math.h>

// ---------------- CDNA5 WMMA types / helpers ----------------
typedef __attribute__((ext_vector_type(16))) __bf16 v16bf;
typedef __attribute__((ext_vector_type(8)))  float  v8f;

#define B_    128
#define L_    64
#define E_    256
#define U_    256
#define KIN   512      // E + U
#define NG    512      // 2U (gate output width)
#define NROWS 16384    // 128 pos + 16256 neg
#define K1    4097
#define K1P   4128     // 129 * 32 (zero-padded K)
#define H1    1024

// A fragment (16x32 bf16) from an LDS tile, row-major stride 32.
// lanes 0-15: row M=lane, K = 0..7 then 16..23; lanes 16-31: row M=lane-16, K = 8..15 then 24..31
static __device__ __forceinline__ v16bf load_a_frag(const __bf16* tile, int lane) {
  const int row  = lane & 15;
  const int koff = (lane >> 4) << 3;
  const __bf16* p = tile + row * 32;
  v16bf a;
#pragma unroll
  for (int e = 0; e < 8; ++e) { a[e] = p[koff + e]; a[8 + e] = p[16 + koff + e]; }
  return a;
}

// B fragment from the pre-swizzled (fragment-order) layout:
// P[((kt*ntiles + nt)*32 + lane)*16 + e]  ->  one aligned 32B vector load per lane
static __device__ __forceinline__ v16bf load_b_packed(const __bf16* __restrict__ P,
                                                      int ntiles, int kt, int nt, int lane) {
  return *(const v16bf*)(P + ((((size_t)kt * ntiles + nt) * 32 + lane) << 4));
}

static __device__ __forceinline__ v8f wmma_bf16(v16bf a, v16bf b, v8f c) {
  return __builtin_amdgcn_wmma_f32_16x16x32_bf16(false, a, false, b, (short)0, c, false, false);
}

static __device__ __forceinline__ float fast_sigmoid(float x) {
  return __builtin_amdgcn_rcpf(1.0f + __expf(-x));     // v_exp + v_rcp
}
static __device__ __forceinline__ float fast_tanh(float x) {
  return 1.0f - 2.0f * __builtin_amdgcn_rcpf(__expf(2.0f * x) + 1.0f);
}

// ---------------- prep kernels ----------------
// Pack nmat stacked [srcK x N] f32 matrices into bf16 fragment-order tiles,
// zero-padding K up to Kpad. dst element i maps to (mat, kt, nt, lane, e).
__global__ void pack_b(const float* __restrict__ src, __bf16* __restrict__ dst,
                       int Kpad, int N, int srcK, int nmat) {
  const int ntiles = N >> 4, ktiles = Kpad >> 5;
  const long total = (long)nmat * Kpad * N;
  for (long i = blockIdx.x * (long)blockDim.x + threadIdx.x; i < total;
       i += gridDim.x * (long)blockDim.x) {
    int e    = (int)(i & 15);
    int lane = (int)((i >> 4) & 31);
    long rest = i >> 9;
    int nt = (int)(rest % ntiles); rest /= ntiles;
    int kt = (int)(rest % ktiles);
    int mat = (int)(rest / ktiles);
    int k = (kt << 5) + ((lane >> 4) << 3) + (e & 7) + ((e >> 3) << 4);
    int n = (nt << 4) + (lane & 15);
    float v = (k < srcK) ? src[(long)mat * srcK * N + (long)k * N + n] : 0.0f;
    dst[i] = (__bf16)v;
  }
}

// xseq[enc][b][t][e]: enc0=qe, enc1=qe reversed, enc2=re, enc3=re reversed (bf16)
__global__ void gather_embed(const int* __restrict__ iq, const int* __restrict__ ir,
                             const int* __restrict__ ql, const int* __restrict__ rl,
                             const float* __restrict__ emb, __bf16* __restrict__ xseq) {
  const int n = 4 * B_ * L_ * E_;
  for (int i = blockIdx.x * blockDim.x + threadIdx.x; i < n; i += gridDim.x * blockDim.x) {
    int e   = i & (E_ - 1);
    int t   = (i >> 8) & (L_ - 1);
    int b   = (i >> 14) & (B_ - 1);
    int enc = i >> 21;
    const int* tok = (enc < 2) ? iq : ir;
    const int* len = (enc < 2) ? ql : rl;
    int tt = t;
    if (enc & 1) { int Lb = len[b]; tt = (t < Lb) ? (Lb - 1 - t) : t; }
    int v = tok[b * L_ + tt];
    xseq[i] = (__bf16)emb[v * E_ + e];
  }
}

// ---------------- GRU step kernels ----------------
// gate: rz = sigmoid([x|h] @ gk + gb); emit rh = bf16(r*h) and z (f32)
__global__ void gru_gate(const __bf16* __restrict__ xseq,
                         const __bf16* __restrict__ h1b, const __bf16* __restrict__ h2b,
                         const float* __restrict__ h1f, const float* __restrict__ h2f,
                         const __bf16* __restrict__ gkb, const float* __restrict__ gb,
                         __bf16* __restrict__ rh, float* __restrict__ zbuf,
                         int t, int layer) {
  const int enc = blockIdx.z;
  const int m0  = blockIdx.x << 4;
  const int tid = threadIdx.x, lane = tid & 31, w = tid >> 5;
  const int ntile = (blockIdx.y << 3) + w;
  const int ncol0 = ntile << 4;

  const __bf16* xa; int xstride;
  if (layer == 0) { xa = xseq + enc * (B_ * L_ * E_) + t * E_; xstride = L_ * E_; }
  else            { xa = h1b + enc * (B_ * U_);                xstride = U_; }
  const __bf16* ha  = ((layer == 0) ? h1b : h2b) + enc * (B_ * U_);
  const float*  hf  = ((layer == 0) ? h1f : h2f) + enc * (B_ * U_);
  const __bf16* gk  = gkb + (size_t)(enc * 2 + layer) * (KIN * NG);  // packed tiles
  const float*  gbb = gb  + (enc * 2 + layer) * NG;

  __shared__ __bf16 Asta[16 * 32];
  v8f acc;
#pragma unroll
  for (int v = 0; v < 8; ++v) acc[v] = 0.0f;

  for (int kc = 0; kc < KIN / 32; ++kc) {
    __syncthreads();
#pragma unroll
    for (int u = 0; u < 2; ++u) {
      int ii = tid * 2 + u;
      int mr = ii >> 5, kk = ii & 31;
      int k = (kc << 5) + kk, m = m0 + mr;
      Asta[ii] = (k < U_) ? xa[m * xstride + k] : ha[m * U_ + (k - U_)];
    }
    __syncthreads();
    v16bf a = load_a_frag(Asta, lane);
    v16bf b = load_b_packed(gk, NG >> 4, kc, ntile, lane);
    acc = wmma_bf16(a, b, acc);
  }

  const int col = ncol0 + (lane & 15);
  const float bias = gbb[col];
#pragma unroll
  for (int v = 0; v < 8; ++v) {
    int row = m0 + v + ((lane >> 4) << 3);
    float s = fast_sigmoid(acc[v] + bias);
    if (col < U_) rh[enc * (B_ * U_) + row * U_ + col]          = (__bf16)(s * hf[row * U_ + col]);
    else          zbuf[enc * (B_ * U_) + row * U_ + (col - U_)] = s;
  }
}

// cand: c = tanh([x|r*h] @ ck + cb); h = mask ? z*h + (1-z)*c : h  (writes f32 + bf16 state)
__global__ void gru_cand(const __bf16* __restrict__ xseq,
                         __bf16* __restrict__ h1b, __bf16* __restrict__ h2b,
                         float* __restrict__ h1f, float* __restrict__ h2f,
                         const __bf16* __restrict__ rh, const float* __restrict__ zbuf,
                         const __bf16* __restrict__ ckb, const float* __restrict__ cb,
                         const int* __restrict__ ql, const int* __restrict__ rl,
                         int t, int layer) {
  const int enc = blockIdx.z;
  const int m0  = blockIdx.x << 4;
  const int tid = threadIdx.x, lane = tid & 31, w = tid >> 5;
  const int ntile = (blockIdx.y << 3) + w;
  const int ncol0 = ntile << 4;

  const __bf16* xa; int xstride;
  if (layer == 0) { xa = xseq + enc * (B_ * L_ * E_) + t * E_; xstride = L_ * E_; }
  else            { xa = h1b + enc * (B_ * U_);                xstride = U_; }
  const __bf16* rhp = rh   + enc * (B_ * U_);
  const float*  zp  = zbuf + enc * (B_ * U_);
  float*  hf = ((layer == 0) ? h1f : h2f) + enc * (B_ * U_);
  __bf16* hb = ((layer == 0) ? h1b : h2b) + enc * (B_ * U_);
  const __bf16* ck  = ckb + (size_t)(enc * 2 + layer) * (KIN * U_);  // packed tiles
  const float*  cbb = cb  + (enc * 2 + layer) * U_;
  const int* len = (enc < 2) ? ql : rl;

  __shared__ __bf16 Asta[16 * 32];
  v8f acc;
#pragma unroll
  for (int v = 0; v < 8; ++v) acc[v] = 0.0f;

  for (int kc = 0; kc < KIN / 32; ++kc) {
    __syncthreads();
#pragma unroll
    for (int u = 0; u < 2; ++u) {
      int ii = tid * 2 + u;
      int mr = ii >> 5, kk = ii & 31;
      int k = (kc << 5) + kk, m = m0 + mr;
      Asta[ii] = (k < U_) ? xa[m * xstride + k] : rhp[m * U_ + (k - U_)];
    }
    __syncthreads();
    v16bf a = load_a_frag(Asta, lane);
    v16bf b = load_b_packed(ck, U_ >> 4, kc, ntile, lane);
    acc = wmma_bf16(a, b, acc);
  }

  const int col = ncol0 + (lane & 15);
  const float cbv = cbb[col];
#pragma unroll
  for (int v = 0; v < 8; ++v) {
    int row = m0 + v + ((lane >> 4) << 3);
    float c  = fast_tanh(acc[v] + cbv);
    float z  = zp[row * U_ + col];
    float ho = hf[row * U_ + col];
    float hn = z * ho + (1.0f - z) * c;
    float o  = (t < len[row]) ? hn : ho;
    hf[row * U_ + col] = o;
    hb[row * U_ + col] = (__bf16)o;
  }
}

// ---------------- epilogue kernels ----------------
__global__ void build_qr(const float* __restrict__ h1f, const float* __restrict__ h2f,
                         float* __restrict__ Qf, float* __restrict__ Rf) {
  const int n = 2 * B_ * 1024;
  for (int i = blockIdx.x * blockDim.x + threadIdx.x; i < n; i += gridDim.x * blockDim.x) {
    int c = i & 1023, b = (i >> 10) & 127, isR = i >> 17;
    int seg = c >> 8, cc = c & 255;
    int enc = isR * 2 + (seg >> 1);
    const float* src = (seg & 1) ? h2f : h1f;
    float v = src[enc * (B_ * U_) + b * U_ + cc];
    (isR ? Rf : Qf)[b * 1024 + c] = v;
  }
}

__global__ void compute_d(const float* __restrict__ Qf, const float* __restrict__ Rf,
                          const int* __restrict__ nqi, const int* __restrict__ nri,
                          float* __restrict__ dvec) {
  int row = blockIdx.x * 8 + (threadIdx.x >> 5);
  int lane = threadIdx.x & 31;
  if (row >= NROWS) return;
  int q, r;
  if (row < B_) { q = row; r = row; }
  else          { q = nqi[row - B_]; r = nri[row - B_]; }
  float s = 0.0f;
  for (int c = lane; c < 1024; c += 32) s += Qf[q * 1024 + c] * Rf[r * 1024 + c];
#pragma unroll
  for (int off = 16; off; off >>= 1) s += __shfl_xor(s, off, 32);
  if (lane == 0) dvec[row] = s;
}

// fused: generate x-tile in LDS -> bf16 WMMA vs packed w1 -> bias+ReLU -> dot w2 -> out
__global__ void mlp_fused(const float* __restrict__ Qf, const float* __restrict__ Rf,
                          const int* __restrict__ nqi, const int* __restrict__ nri,
                          const float* __restrict__ dvec, const __bf16* __restrict__ w1p,
                          const float* __restrict__ b1, const float* __restrict__ w2,
                          const float* __restrict__ b2, float* __restrict__ out) {
  const int m0  = blockIdx.x << 4;
  const int tid = threadIdx.x, lane = tid & 31, w = tid >> 5;
  __shared__ __bf16 Asta[16 * 32];
  __shared__ float oacc[16][2];
  __shared__ int qidx[16], ridx[16];
  __shared__ float dsh[16];
  if (tid < 16) {
    int row = m0 + tid, q, r;
    if (row < B_) { q = row; r = row; }
    else          { q = nqi[row - B_]; r = nri[row - B_]; }
    qidx[tid] = q; ridx[tid] = r; dsh[tid] = dvec[row];
  }
  if (tid < 32) oacc[tid >> 1][tid & 1] = 0.0f;

  v8f acc[8];
#pragma unroll
  for (int j = 0; j < 8; ++j)
#pragma unroll
    for (int v = 0; v < 8; ++v) acc[j][v] = 0.0f;

  for (int kc = 0; kc < K1P / 32; ++kc) {   // 129 K-steps
    __syncthreads();
#pragma unroll
    for (int u = 0; u < 2; ++u) {
      int ii = tid * 2 + u;
      int mr = ii >> 5, kk = ii & 31;
      int c = (kc << 5) + kk;
      int q = qidx[mr], r = ridx[mr];
      float v;
      if      (c < 1024)  v = Qf[q * 1024 + c];
      else if (c == 1024) v = dsh[mr];
      else if (c < 2049)  { int cc = c - 1025; v = Qf[q * 1024 + cc] - Rf[r * 1024 + cc]; }
      else if (c < 3073)  { int cc = c - 2049; v = Qf[q * 1024 + cc] * Rf[r * 1024 + cc]; }
      else if (c < 4097)  { int cc = c - 3073; v = Rf[r * 1024 + cc]; }
      else                v = 0.0f;
      Asta[ii] = (__bf16)v;
    }
    __syncthreads();
    v16bf a = load_a_frag(Asta, lane);
#pragma unroll
    for (int j = 0; j < 8; ++j) {
      v16bf b = load_b_packed(w1p, H1 >> 4, kc, (w << 3) + j, lane);
      acc[j] = wmma_bf16(a, b, acc[j]);
    }
  }

  // bias + ReLU + (h @ w2) via wave shuffle reduction over the 16-lane N halves
#pragma unroll
  for (int j = 0; j < 8; ++j) {
    int col = (((w << 3) + j) << 4) + (lane & 15);
    float w20 = w2[col * 2 + 0], w21 = w2[col * 2 + 1];
    float bb = b1[col];
#pragma unroll
    for (int v = 0; v < 8; ++v) {
      float h = fmaxf(acc[j][v] + bb, 0.0f);
      float p0 = h * w20, p1 = h * w21;
#pragma unroll
      for (int off = 8; off; off >>= 1) {
        p0 += __shfl_xor(p0, off, 32);
        p1 += __shfl_xor(p1, off, 32);
      }
      if ((lane & 15) == 0) {
        int rl = v + ((lane >> 4) << 3);
        atomicAdd(&oacc[rl][0], p0);
        atomicAdd(&oacc[rl][1], p1);
      }
    }
  }
  __syncthreads();
  if (tid < 32) {
    int rl = tid >> 1, jj = tid & 1;
    out[(m0 + rl) * 2 + jj] = oacc[rl][jj] + b2[jj];
  }
}

// ---------------- host launcher ----------------
extern "C" void kernel_launch(void* const* d_in, const int* in_sizes, int n_in,
                              void* d_out, int out_size, void* d_ws, size_t ws_size,
                              hipStream_t stream) {
  const int*   iq  = (const int*)d_in[0];
  const int*   ir  = (const int*)d_in[1];
  const int*   ql  = (const int*)d_in[2];
  const int*   rl  = (const int*)d_in[3];
  const int*   nqi = (const int*)d_in[4];
  const int*   nri = (const int*)d_in[5];
  const float* emb = (const float*)d_in[6];
  const float* gk  = (const float*)d_in[7];
  const float* gb  = (const float*)d_in[8];
  const float* ck  = (const float*)d_in[9];
  const float* cb  = (const float*)d_in[10];
  const float* w1  = (const float*)d_in[11];
  const float* b1  = (const float*)d_in[12];
  const float* w2  = (const float*)d_in[13];
  const float* b2  = (const float*)d_in[14];
  float* out = (float*)d_out;

  char* p = (char*)d_ws;
  auto alloc = [&](size_t bytes) { char* r = p; p += (bytes + 255) & ~size_t(255); return r; };
  __bf16* xseq = (__bf16*)alloc((size_t)4 * B_ * L_ * E_ * 2);
  __bf16* gkb  = (__bf16*)alloc((size_t)8 * KIN * NG * 2);     // packed
  __bf16* ckb  = (__bf16*)alloc((size_t)8 * KIN * U_ * 2);     // packed
  __bf16* w1p  = (__bf16*)alloc((size_t)K1P * H1 * 2);         // packed, zero-padded K
  float*  h1f  = (float*)alloc((size_t)4 * B_ * U_ * 4);
  float*  h2f  = (float*)alloc((size_t)4 * B_ * U_ * 4);
  __bf16* h1b  = (__bf16*)alloc((size_t)4 * B_ * U_ * 2);
  __bf16* h2b  = (__bf16*)alloc((size_t)4 * B_ * U_ * 2);
  __bf16* rh   = (__bf16*)alloc((size_t)4 * B_ * U_ * 2);
  float*  zbuf = (float*)alloc((size_t)4 * B_ * U_ * 4);
  float*  Qf   = (float*)alloc((size_t)B_ * 1024 * 4);
  float*  Rf   = (float*)alloc((size_t)B_ * 1024 * 4);
  float*  dvec = (float*)alloc((size_t)NROWS * 4);

  hipMemsetAsync(h1f, 0, (size_t)4 * B_ * U_ * 4, stream);
  hipMemsetAsync(h2f, 0, (size_t)4 * B_ * U_ * 4, stream);
  hipMemsetAsync(h1b, 0, (size_t)4 * B_ * U_ * 2, stream);
  hipMemsetAsync(h2b, 0, (size_t)4 * B_ * U_ * 2, stream);

  pack_b<<<512, 256, 0, stream>>>(gk, gkb, KIN, NG, KIN, 8);
  pack_b<<<512, 256, 0, stream>>>(ck, ckb, KIN, U_, KIN, 8);
  pack_b<<<1024, 256, 0, stream>>>(w1, w1p, K1P, H1, K1, 1);
  gather_embed<<<2048, 256, 0, stream>>>(iq, ir, ql, rl, emb, xseq);

  dim3 gGate(8, 4, 4), gCand(8, 2, 4);
  for (int t = 0; t < L_; ++t) {
    for (int layer = 0; layer < 2; ++layer) {
      gru_gate<<<gGate, 256, 0, stream>>>(xseq, h1b, h2b, h1f, h2f, gkb, gb, rh, zbuf, t, layer);
      gru_cand<<<gCand, 256, 0, stream>>>(xseq, h1b, h2b, h1f, h2f, rh, zbuf, ckb, cb, ql, rl, t, layer);
    }
  }

  build_qr<<<512, 256, 0, stream>>>(h1f, h2f, Qf, Rf);
  compute_d<<<NROWS / 8, 256, 0, stream>>>(Qf, Rf, nqi, nri, dvec);
  mlp_fused<<<NROWS / 16, 256, 0, stream>>>(Qf, Rf, nqi, nri, dvec, w1p, b1, w2, b2, out);

  (void)in_sizes; (void)n_in; (void)out_size; (void)ws_size;
}
